// minLSTM_29394756174349
// MI455X (gfx1250) — matrix operand order repaired and checked by reading, not compiled, not run on hardware
//
#include <hip/hip_runtime.h>

// ---------------------------------------------------------------------------
// minLSTM fused forward for MI455X (gfx1250, wave32, WMMA bf16)
//   x:       [8, 8192, 256] fp32
//   w_gates: [768, 256]     fp32   (rows: hidden | f_gate | i_gate, 256 each)
//   out:     [8, 8192, 256] fp32
// 3-pass chunked scan: (1) GEMM+local scan -> chunk carries, (2) scan carries,
// (3) recompute GEMM+local scan, apply prefix, exp, store.
// Weights (and x, if ws_size allows) are pre-converted to bf16 once so the hot
// GEMM kernels do (almost) no conversion VALU work; all log-space math uses
// fast-path transcendentals (v_exp_f32 / v_log_f32 only, no libm log1pf).
// ---------------------------------------------------------------------------

typedef __attribute__((ext_vector_type(16))) __bf16 v16bf;
typedef __attribute__((ext_vector_type(8)))  __bf16 v8bf;
typedef __attribute__((ext_vector_type(8)))  float  v8f;
typedef __attribute__((ext_vector_type(8)))  float  v8fl;

#define DIM      256
#define SEQ      8192
#define NBATCH   8
#define CHUNK    128                 // seq rows per block
#define NCHUNK   (SEQ / CHUNK)       // 64
#define DTILES   (DIM / 16)          // 16
#define NEG_INF  (-1.0e30f)

#define NCARRY   ((long)NBATCH * NCHUNK * DIM)         // 131072
#define WS_CARRY_BYTES (3 * NCARRY * 4)                // 1.5 MB
#define WBF_OFF  WS_CARRY_BYTES                        // bf16 weights
#define WBF_BYTES ((long)3 * DIM * DIM * 2)            // 384 KB
#define XBF_OFF  (WBF_OFF + WBF_BYTES)
#define XBF_BYTES ((long)NBATCH * SEQ * DIM * 2)       // 33.5 MB
#define WS_FULL  (XBF_OFF + XBF_BYTES)

// log(1+t) for t in (0,1]: plain log is accurate enough here (abs err ~1e-7,
// well under the bf16-GEMM noise floor) and avoids the libm log1pf expansion.
static __device__ __forceinline__ float log1p_fast(float t) {
    return __logf(1.0f + t);
}

static __device__ __forceinline__ float softplusf(float z) {
    // log(1 + e^z), numerically safe: max(z,0) + log(1 + e^-|z|)
    return fmaxf(z, 0.0f) + log1p_fast(__expf(-fabsf(z)));
}

static __device__ __forceinline__ float lae(float xx, float yy) {
    // logaddexp with -1e30 acting as -inf sentinel (exp(-huge)=0 -> m)
    float m = fmaxf(xx, yy);
    float n = fminf(xx, yy);
    return m + log1p_fast(__expf(n - m));
}

// -------------------- one-shot fp32 -> bf16 conversions ---------------------
__global__ __launch_bounds__(256) void convert_f32_bf16(
    const float* __restrict__ src, __bf16* __restrict__ dst, long n8)
{
    long i = ((long)blockIdx.x * blockDim.x + threadIdx.x);
    if (i >= n8) return;
    v8fl v = *(const v8fl*)(src + i * 8);
    *(v8bf*)(dst + i * 8) = __builtin_convertvector(v, v8bf);
}

// PASS == 1: write per-chunk carries (A = sum log_f, B = local log_h end)
// PASS == 3: read chunk prefix, apply, exp, store output
// XBF: x pre-converted to bf16 in workspace
template <int PASS, bool XBF>
__global__ __launch_bounds__(256)
void minlstm_fused(const float* __restrict__ x, const __bf16* __restrict__ xbf,
                   const __bf16* __restrict__ wbf,
                   float* __restrict__ out,
                   float* __restrict__ carryA, float* __restrict__ carryB,
                   const float* __restrict__ prefixB)
{
    __shared__ float ldsA[8][16];
    __shared__ float ldsB[8][16];

    const int tid   = threadIdx.x;
    const int lane  = tid & 31;
    const int w     = tid >> 5;        // wave id 0..7 -> 16-row M tile
    const int col   = lane & 15;       // output channel within d-tile
    const int khalf = lane >> 4;       // 0: rows 0-7 / K-lo, 1: rows 8-15 / K-hi

    const int blk = blockIdx.x;        // 8 * 64 * 16 = 8192 blocks
    const int dt  = blk & 15;
    const int ch  = (blk >> 4) & (NCHUNK - 1);
    const int b   = blk >> 10;

    const int  d0      = dt * 16;
    const long rowBase = (long)b * SEQ + (long)ch * CHUNK + w * 16;

    // ---------------- tri-GEMM: 16x16 tile per gate, K = 256 ----------------
    v8f accH = {}, accF = {}, accI = {};
    const float*  xrow  = x   + (rowBase + col) * DIM;            // A row (M = col)
    const __bf16* xrowb = xbf + (rowBase + col) * DIM;
    const __bf16* wH    = wbf + (long)(0 * DIM + d0 + col) * DIM; // B col (N = col)
    const __bf16* wF    = wbf + (long)(1 * DIM + d0 + col) * DIM;
    const __bf16* wI    = wbf + (long)(2 * DIM + d0 + col) * DIM;

#pragma unroll
    for (int kk = 0; kk < DIM; kk += 32) {
        // A fragment: contiguous k order (same as B) — lanes 0-15: K 0..15,
        // lanes 16-31: K 16..31
        v8bf A0, A1;
        if (XBF) {
            A0 = *(const v8bf*)(xrowb + kk + khalf * 16);
            A1 = *(const v8bf*)(xrowb + kk + 8 + khalf * 16);
        } else {
            if (kk + 32 < DIM) __builtin_prefetch(xrow + kk + 32, 0, 0);
            A0 = __builtin_convertvector(*(const v8fl*)(xrow + kk + khalf * 16), v8bf);
            A1 = __builtin_convertvector(*(const v8fl*)(xrow + kk + 8 + khalf * 16), v8bf);
        }
        v16bf A = __builtin_shufflevector(A0, A1, 0, 1, 2, 3, 4, 5, 6, 7,
                                                  8, 9, 10, 11, 12, 13, 14, 15);

        // B fragments: contiguous k order as in A — lanes 0-15: K 0..15, lanes 16-31: K 16..31
        const int kb = kk + khalf * 16;
        v16bf BH = *(const v16bf*)(wH + kb);
        v16bf BF = *(const v16bf*)(wF + kb);
        v16bf BI = *(const v16bf*)(wI + kb);

        accH = __builtin_amdgcn_wmma_f32_16x16x32_bf16(false, A, false, BH,
                                                       (short)0, accH, false, false);
        accF = __builtin_amdgcn_wmma_f32_16x16x32_bf16(false, A, false, BF,
                                                       (short)0, accF, false, false);
        accI = __builtin_amdgcn_wmma_f32_16x16x32_bf16(false, A, false, BI,
                                                       (short)0, accI, false, false);
    }

    // -------- fused gate math (C layout: vgpr r = row khalf*8+r, N = col) ----
    float lf[8], lv[8];
#pragma unroll
    for (int r = 0; r < 8; ++r) {
        float h  = accH[r];
        float fg = accF[r];
        float ig = accI[r];
        float diff = softplusf(-fg) - softplusf(-ig);
        float spd  = softplusf(diff);
        float logf_ = -spd;                       // log f' = -softplus(diff)
        float logi_ = diff - spd;                 // log i' = -softplus(-diff)
        float logg  = (h >= 0.0f) ? __logf(h + 0.5f) : -softplusf(-h);
        lf[r] = logf_;
        lv[r] = logi_ + logg;                     // log(i' * g(h~))
    }

    // -------- local scan: log_h[t] = logaddexp(lf[t] + log_h[t-1], lv[t]) ----
    float aL[8], bL[8];
    float a = 0.0f, bs = NEG_INF;
#pragma unroll
    for (int r = 0; r < 8; ++r) {
        a  += lf[r];
        bs  = lae(lf[r] + bs, lv[r]);
        aL[r] = a; bL[r] = bs;
    }
    // stitch rows 0-7 (lanes 0-15) into rows 8-15 (lanes 16-31)
    float Alo = __shfl(a,  col, 32);
    float Blo = __shfl(bs, col, 32);
    if (khalf) {
#pragma unroll
        for (int r = 0; r < 8; ++r) {
            bL[r] = lae(aL[r] + Blo, bL[r]);
            aL[r] += Alo;
        }
    }
    // tile aggregates (row 15) -> LDS for cross-wave combine
    if (khalf) { ldsA[w][col] = aL[7]; ldsB[w][col] = bL[7]; }
    __syncthreads();

    const long ci = ((long)b * NCHUNK + ch) * DIM + d0 + col;

    if (PASS == 1) {
        if (w == 0 && khalf == 0) {
            float Ac = 0.0f, Bc = NEG_INF;
#pragma unroll
            for (int t = 0; t < 8; ++t) {
                Bc  = lae(ldsA[t][col] + Bc, ldsB[t][col]);
                Ac += ldsA[t][col];
            }
            carryA[ci] = Ac;
            carryB[ci] = Bc;
        }
    } else {
        // prefix state entering this wave's tile (includes chunk prefix)
        float Bin = prefixB[ci];
        for (int t = 0; t < w; ++t)
            Bin = lae(ldsA[t][col] + Bin, ldsB[t][col]);
#pragma unroll
        for (int r = 0; r < 8; ++r) {
            float logh = lae(aL[r] + Bin, bL[r]);
            out[(rowBase + khalf * 8 + r) * DIM + d0 + col] = __expf(logh);
        }
    }
}

// ------------- pass 2: scan the 64 chunk carries per (batch, channel) -------
__global__ __launch_bounds__(256)
void minlstm_scan_carries(const float* __restrict__ carryA,
                          const float* __restrict__ carryB,
                          float* __restrict__ prefixB)
{
    int t = blockIdx.x * blockDim.x + threadIdx.x;   // 0 .. 2047
    if (t >= NBATCH * DIM) return;
    int b = t >> 8, d = t & (DIM - 1);
    float B = NEG_INF;
    for (int c = 0; c < NCHUNK; ++c) {
        long i = ((long)b * NCHUNK + c) * DIM + d;
        prefixB[i] = B;                               // state BEFORE chunk c
        B = lae(carryA[i] + B, carryB[i]);
    }
}

extern "C" void kernel_launch(void* const* d_in, const int* in_sizes, int n_in,
                              void* d_out, int out_size, void* d_ws, size_t ws_size,
                              hipStream_t stream) {
    const float* x  = (const float*)d_in[0];
    const float* wg = (const float*)d_in[1];
    float* out = (float*)d_out;

    char* ws = (char*)d_ws;
    float*  carryA  = (float*)ws;
    float*  carryB  = carryA + NCARRY;
    float*  prefixB = carryB + NCARRY;
    __bf16* wbf     = (__bf16*)(ws + WBF_OFF);
    __bf16* xbf     = (__bf16*)(ws + XBF_OFF);

    // Convert weights once: 768*256 / 8 = 24576 vector threads
    {
        long n8 = (long)3 * DIM * DIM / 8;
        convert_f32_bf16<<<(int)((n8 + 255) / 256), 256, 0, stream>>>(wg, wbf, n8);
    }

    const int  nblocks = NBATCH * NCHUNK * DTILES;    // 8192
    const bool bigWs   = ws_size >= (size_t)WS_FULL;  // room for bf16 x?

    if (bigWs) {
        long n8 = (long)NBATCH * SEQ * DIM / 8;       // 2,097,152
        convert_f32_bf16<<<(int)((n8 + 255) / 256), 256, 0, stream>>>(x, xbf, n8);
        minlstm_fused<1, true><<<nblocks, 256, 0, stream>>>(
            x, xbf, wbf, out, carryA, carryB, prefixB);
        minlstm_scan_carries<<<(NBATCH * DIM + 255) / 256, 256, 0, stream>>>(
            carryA, carryB, prefixB);
        minlstm_fused<3, true><<<nblocks, 256, 0, stream>>>(
            x, xbf, wbf, out, carryA, carryB, prefixB);
    } else {
        minlstm_fused<1, false><<<nblocks, 256, 0, stream>>>(
            x, xbf, wbf, out, carryA, carryB, prefixB);
        minlstm_scan_carries<<<(NBATCH * DIM + 255) / 256, 256, 0, stream>>>(
            carryA, carryB, prefixB);
        minlstm_fused<3, false><<<nblocks, 256, 0, stream>>>(
            x, xbf, wbf, out, carryA, carryB, prefixB);
    }
}
